// BertSelfAttention_11304353923397
// MI455X (gfx1250) — matrix operand order repaired and checked by reading, not compiled, not run on hardware
//
#include <hip/hip_runtime.h>

// ---------------------------------------------------------------------------
// Types for CDNA5 WMMA (wave32): bf16 A/B fragments (16 halves/lane = 8 VGPRs),
// f32 C/D accumulator (8 floats/lane = 8 VGPRs).
// ---------------------------------------------------------------------------
typedef __attribute__((ext_vector_type(8)))  float  v8f;
typedef __attribute__((ext_vector_type(16))) __bf16 v16bf;
typedef __attribute__((ext_vector_type(8)))  __bf16 v8bf;

#define WMMA_BF16(A, B, C) \
    __builtin_amdgcn_wmma_f32_16x16x32_bf16(false, (A), false, (B), (short)0, (C), false, false)

static __device__ inline v8f vzero8f() {
    v8f z;
#pragma unroll
    for (int i = 0; i < 8; ++i) z[i] = 0.0f;
    return z;
}
static __device__ inline v8bf vzero8bf() {
    v8bf z;
#pragma unroll
    for (int i = 0; i < 8; ++i) z[i] = (__bf16)0.0f;
    return z;
}
static __device__ inline v16bf join8(v8bf lo, v8bf hi) {
    v16bf r;
#pragma unroll
    for (int i = 0; i < 8; ++i) { r[i] = lo[i]; r[i + 8] = hi[i]; }
    return r;
}

// 16-byte async copy global -> LDS (CDNA5, tracked by ASYNCcnt).
static __device__ inline void async_copy16(unsigned lds_off, const void* gptr) {
    asm volatile("global_load_async_to_lds_b128 %0, %1, off"
                 :: "v"(lds_off), "v"(gptr) : "memory");
}

// ---------------------------------------------------------------------------
// Problem constants
// ---------------------------------------------------------------------------
#define BATCH   2
#define SEQ     2048
#define HID     1024
#define NHEAD   16
#define HDIM    64
#define MAXPOS  2048
#define NDIST   (2 * MAXPOS - 1)   // 4095 rows in dist_emb

// ---------------------------------------------------------------------------
// Kernel 1: fp32 -> bf16 elementwise convert
// ---------------------------------------------------------------------------
__global__ void cvt_bf16_kernel(const float* __restrict__ src,
                                __bf16* __restrict__ dst, int n) {
    int i = blockIdx.x * blockDim.x + threadIdx.x;
    if (i < n) dst[i] = (__bf16)src[i];
}

// ---------------------------------------------------------------------------
// Kernel 2: projection GEMM  C[m,n] = sum_f A[m,f]*W[n,f] + bias[n]
//   dst (trans=0): [B,nh,S,hd] bf16 head-major (Q, K)
//   dst (trans=1): [B,nh,hd,S] bf16 transposed (V -> key index contiguous)
// Block = 128 threads (4 waves). Tile 64(M) x 64(N); wave w owns N-sub w.
// ---------------------------------------------------------------------------
__global__ __launch_bounds__(128)
void proj_kernel(const __bf16* __restrict__ A, const __bf16* __restrict__ W,
                 const float* __restrict__ bias, __bf16* __restrict__ dst,
                 int trans) {
    const int lane = threadIdx.x & 31;
    const int wave = threadIdx.x >> 5;
    const int half = lane >> 4;
    const int lid  = lane & 15;

    const int m0 = blockIdx.x * 64;
    const int n0 = blockIdx.y * 64 + wave * 16;

    v8f acc[4];
#pragma unroll
    for (int t = 0; t < 4; ++t) acc[t] = vzero8f();

    const __bf16* wrow = W + (size_t)(n0 + lid) * HID;

    for (int ks = 0; ks < HID; ks += 32) {
        v16bf bw = *(const v16bf*)(wrow + ks + 16 * half);
#pragma unroll
        for (int mt = 0; mt < 4; ++mt) {
            const __bf16* ar = A + (size_t)(m0 + mt * 16 + lid) * HID + ks + 16 * half;
            v16bf af = join8(*(const v8bf*)ar, *(const v8bf*)(ar + 8));
            acc[mt] = WMMA_BF16(af, bw, acc[mt]);
        }
    }

    const int   n  = n0 + lid;
    const float bv = bias[n];
    const int   hh = n >> 6;       // head
    const int   d  = n & 63;       // dim within head
#pragma unroll
    for (int mt = 0; mt < 4; ++mt) {
#pragma unroll
        for (int r = 0; r < 8; ++r) {
            int m    = m0 + mt * 16 + r + 8 * half;
            int b    = m >> 11;
            int sidx = m & (SEQ - 1);
            float v  = acc[mt][r] + bv;
            size_t idx;
            if (trans == 0)
                idx = (((size_t)(b * NHEAD + hh) * SEQ + sidx) * HDIM) + d;
            else
                idx = (((size_t)(b * NHEAD + hh) * HDIM + d) * SEQ) + sidx;
            dst[idx] = (__bf16)v;
        }
    }
}

// ---------------------------------------------------------------------------
// Kernel 3: flash attention with relative-position scores.
// Block = 128 threads (4 waves), all sharing one (b,h); wave w owns q rows
// [q0, q0+16). K/V 16-key tiles are staged in LDS via async global->LDS
// copies (double-buffered, ASYNCcnt) and shared by all 4 waves.
// Per tile per wave: 2 WMMA (Q*K^T) + 8 WMMA (rel bands) + 4 WMMA (P*V).
// ---------------------------------------------------------------------------
__global__ __launch_bounds__(128)
void flash_rel_kernel(const __bf16* __restrict__ Q, const __bf16* __restrict__ K,
                      const __bf16* __restrict__ Vt, const __bf16* __restrict__ E,
                      float* __restrict__ out) {
    const int lane = threadIdx.x & 31;
    const int wave = threadIdx.x >> 5;
    const int half = lane >> 4;
    const int lid  = lane & 15;
    const int tid  = threadIdx.x;

    const int qblk = blockIdx.x & ((SEQ / 64) - 1);
    const int bh   = blockIdx.x / (SEQ / 64);
    const int b    = bh >> 4;
    const int h    = bh & 15;
    const int q0   = qblk * 64 + wave * 16;

    const __bf16* Qb = Q  + (size_t)(b * NHEAD + h) * SEQ * HDIM;
    const __bf16* Kb = K  + (size_t)(b * NHEAD + h) * SEQ * HDIM;
    const __bf16* Vb = Vt + (size_t)(b * NHEAD + h) * HDIM * SEQ;

    // K tile: 16 rows x 64 halves (row-major, contiguous 2KB in global).
    // V tile: 64 d-rows x 16 keys (32B per row).
    __shared__ __attribute__((aligned(32))) __bf16 ldsK[2][16 * 64];
    __shared__ __attribute__((aligned(32))) __bf16 ldsV[2][64 * 16];
    __shared__ float  lds_rq[4][16 * 32];
    __shared__ float  lds_rk[4][16 * 32];
    __shared__ __attribute__((aligned(32))) __bf16 lds_p[4][16 * 16];
    float*  RQ = lds_rq[wave];
    float*  RK = lds_rk[wave];
    __bf16* P  = lds_p [wave];

    // Issue async copies of one K/V tile into LDS buffer bi (128 threads,
    // each thread moves 16B of K and 16B of V).
    auto issue_tile = [&](int k0, int bi) {
        unsigned kbase = (unsigned)(uintptr_t)(&ldsK[bi][0]);
        unsigned vbase = (unsigned)(uintptr_t)(&ldsV[bi][0]);
        async_copy16(kbase + tid * 16,
                     (const char*)(Kb + (size_t)k0 * HDIM) + tid * 16);
        int d = tid >> 1, sub = tid & 1;
        async_copy16(vbase + d * 32 + sub * 16,
                     Vb + (size_t)d * SEQ + k0 + sub * 8);
    };

    // Q A-fragments (constant over the k loop)
    v16bf aq[2];
#pragma unroll
    for (int s = 0; s < 2; ++s) {
        const __bf16* qr = Qb + (size_t)(q0 + lid) * HDIM + s * 32 + 16 * half;
        aq[s] = join8(*(const v8bf*)qr, *(const v8bf*)(qr + 8));
    }

    float mrow[8], lrow[8];
    v8f o[4];
#pragma unroll
    for (int r = 0; r < 8; ++r) { mrow[r] = -1e30f; lrow[r] = 0.0f; }
#pragma unroll
    for (int t = 0; t < 4; ++t) o[t] = vzero8f();

    issue_tile(0, 0);   // prologue: tile 0 into buffer 0

    for (int k0 = 0; k0 < SEQ; k0 += 16) {
        const int cur = (k0 >> 4) & 1;
        __syncthreads();   // everyone done reading buffer cur^1 (prev iter)
        if (k0 + 16 < SEQ) {
            issue_tile(k0 + 16, cur ^ 1);
            // async loads complete in order: cnt<=2 => current tile landed
            asm volatile("s_wait_asynccnt 2" ::: "memory");
        } else {
            asm volatile("s_wait_asynccnt 0" ::: "memory");
        }
        __syncthreads();   // all waves' async copies of `cur` visible
        const __bf16* Kt = ldsK[cur];
        const __bf16* Vl = ldsV[cur];

        // ---- scores tile Q*K^T and K A-fragments (for rel_k), from LDS ----
        v8f sa = vzero8f();
        v16bf ak[2];
#pragma unroll
        for (int s = 0; s < 2; ++s) {
            const __bf16* kr = Kt + lid * HDIM + s * 32;
            v16bf bk = *(const v16bf*)(kr + 16 * half);
            ak[s] = join8(*(const v8bf*)(kr + 16 * half),
                          *(const v8bf*)(kr + 16 * half + 8));
            sa = WMMA_BF16(aq[s], bk, sa);
        }

        // ---- relative-position bands: RQ = Q*E^T, RK = K*E^T (16x32) ----
        const int j0 = (MAXPOS - 1) + q0 - k0 - 15;
#pragma unroll
        for (int ns = 0; ns < 2; ++ns) {
            int j = j0 + ns * 16 + lid;
            if (j < 0) j = 0;
            if (j > NDIST - 1) j = NDIST - 1;   // only col 31 (unused) clamps
            v8f rq = vzero8f(), rk = vzero8f();
#pragma unroll
            for (int s = 0; s < 2; ++s) {
                v16bf eb = *(const v16bf*)(E + (size_t)j * HDIM + s * 32 + 16 * half);
                rq = WMMA_BF16(aq[s], eb, rq);
                rk = WMMA_BF16(ak[s], eb, rk);
            }
#pragma unroll
            for (int r = 0; r < 8; ++r) {
                int M = r + 8 * half;
                RQ[M * 32 + ns * 16 + lid] = rq[r];
                RK[M * 32 + ns * 16 + lid] = rk[r];
            }
        }
        asm volatile("s_wait_dscnt 0" ::: "memory");  // same-wave LDS RAW

        // ---- gather band + scale, online softmax ----
        float vals[8];
#pragma unroll
        for (int r = 0; r < 8; ++r) {
            int M = r + 8 * half;
            int N = lid;
            int c = M - N + 15;                  // 0..30
            vals[r] = (sa[r] + RQ[M * 32 + c] + RK[N * 32 + c]) * 0.125f;
        }
        float fac[8], p[8];
#pragma unroll
        for (int r = 0; r < 8; ++r) {
            float v = vals[r];
#pragma unroll
            for (int off = 1; off < 16; off <<= 1)
                v = fmaxf(v, __shfl_xor(v, off, 16));
            float mn = fmaxf(mrow[r], v);
            float pv = __expf(vals[r] - mn);
            float su = pv;
#pragma unroll
            for (int off = 1; off < 16; off <<= 1)
                su += __shfl_xor(su, off, 16);
            fac[r]  = __expf(mrow[r] - mn);
            lrow[r] = lrow[r] * fac[r] + su;
            mrow[r] = mn;
            p[r]    = pv;
        }
#pragma unroll
        for (int t = 0; t < 4; ++t)
#pragma unroll
            for (int r = 0; r < 8; ++r) o[t][r] *= fac[r];

        // ---- P through LDS as bf16, then O += P*V ----
#pragma unroll
        for (int r = 0; r < 8; ++r) {
            int M = r + 8 * half;
            P[M * 16 + lid] = (__bf16)p[r];
        }
        asm volatile("s_wait_dscnt 0" ::: "memory");

        v16bf pa = (half == 0) ? *(const v16bf*)(P + lid * 16)
                               : join8(vzero8bf(), vzero8bf());       // keys 16..31 padded zero
#pragma unroll
        for (int t = 0; t < 4; ++t) {
            v16bf bv;
            if (half == 0) {                      // B rows 0..15 (valid keys)
                bv = *(const v16bf*)(Vl + (t * 16 + lid) * 16);
            } else {                              // B rows 16..31 -> zero pad
                bv = join8(vzero8bf(), vzero8bf());
            }
            o[t] = WMMA_BF16(pa, bv, o[t]);
        }
    }

    // ---- epilogue: normalize and write ctx [B,S,1024] fp32 ----
    const size_t obase = (size_t)b * SEQ * HID + (size_t)h * HDIM;
#pragma unroll
    for (int r = 0; r < 8; ++r) {
        int   M   = r + 8 * half;
        int   q   = q0 + M;
        float inv = 1.0f / lrow[r];
#pragma unroll
        for (int t = 0; t < 4; ++t)
            out[obase + (size_t)q * HID + t * 16 + lid] = o[t][r] * inv;
    }
}

// ---------------------------------------------------------------------------
// Host-side launcher
// ---------------------------------------------------------------------------
extern "C" void kernel_launch(void* const* d_in, const int* in_sizes, int n_in,
                              void* d_out, int out_size, void* d_ws, size_t ws_size,
                              hipStream_t stream) {
    const float* hs = (const float*)d_in[0];
    const float* Wq = (const float*)d_in[1];
    const float* bq = (const float*)d_in[2];
    const float* Wk = (const float*)d_in[3];
    const float* bk = (const float*)d_in[4];
    const float* Wv = (const float*)d_in[5];
    const float* bv = (const float*)d_in[6];
    const float* de = (const float*)d_in[7];

    const int n_hs = BATCH * SEQ * HID;          // 4194304
    const int n_w  = HID * HID;                  // 1048576
    const int n_e  = NDIST * HDIM;               // 262080
    const int n_qk = BATCH * NHEAD * SEQ * HDIM; // 4194304

    char* ws = (char*)d_ws;
    size_t off = 0;
    auto carve = [&](size_t bytes) { char* p = ws + off; off += (bytes + 255) & ~(size_t)255; return p; };
    __bf16* hs_bf = (__bf16*)carve((size_t)n_hs * 2);
    __bf16* wq_bf = (__bf16*)carve((size_t)n_w * 2);
    __bf16* wk_bf = (__bf16*)carve((size_t)n_w * 2);
    __bf16* wv_bf = (__bf16*)carve((size_t)n_w * 2);
    __bf16* e_bf  = (__bf16*)carve((size_t)n_e * 2);
    __bf16* q_bf  = (__bf16*)carve((size_t)n_qk * 2);
    __bf16* k_bf  = (__bf16*)carve((size_t)n_qk * 2);
    __bf16* vt_bf = (__bf16*)carve((size_t)n_qk * 2);

    // fp32 -> bf16 conversions
    cvt_bf16_kernel<<<(n_hs + 255) / 256, 256, 0, stream>>>(hs, hs_bf, n_hs);
    cvt_bf16_kernel<<<(n_w  + 255) / 256, 256, 0, stream>>>(Wq, wq_bf, n_w);
    cvt_bf16_kernel<<<(n_w  + 255) / 256, 256, 0, stream>>>(Wk, wk_bf, n_w);
    cvt_bf16_kernel<<<(n_w  + 255) / 256, 256, 0, stream>>>(Wv, wv_bf, n_w);
    cvt_bf16_kernel<<<(n_e  + 255) / 256, 256, 0, stream>>>(de, e_bf,  n_e);

    // WMMA projections: Q, K head-major; V transposed
    dim3 pgrid(BATCH * SEQ / 64, HID / 64);
    proj_kernel<<<pgrid, 128, 0, stream>>>(hs_bf, wq_bf, bq, q_bf,  0);
    proj_kernel<<<pgrid, 128, 0, stream>>>(hs_bf, wk_bf, bk, k_bf,  0);
    proj_kernel<<<pgrid, 128, 0, stream>>>(hs_bf, wv_bf, bv, vt_bf, 1);

    // flash attention with relative position
    flash_rel_kernel<<<BATCH * NHEAD * (SEQ / 64), 128, 0, stream>>>(
        q_bf, k_bf, vt_bf, e_bf, (float*)d_out);
}
